// Block_35820027249020
// MI455X (gfx1250) — matrix operand order repaired and checked by reading, not compiled, not run on hardware
//
#include <hip/hip_runtime.h>

typedef _Float16 half16 __attribute__((ext_vector_type(16)));
typedef float    float8 __attribute__((ext_vector_type(8)));

#define EPSV 1e-5f
#define MTOT 8192   // T*B*H*W = 4*8*256

__device__ __forceinline__ float8 f8zero() {
    float8 z = {0.f,0.f,0.f,0.f,0.f,0.f,0.f,0.f};
    return z;
}

// Vectorized WMMA fragment load: two aligned 16B chunks -> 16 halves.
__device__ __forceinline__ half16 ld_frag(const _Float16* p0, const _Float16* p1) {
    union { half16 v; uint4 q[2]; } u;
    u.q[0] = *(const uint4*)p0;
    u.q[1] = *(const uint4*)p1;
    return u.v;
}

// ---------------------------------------------------------------------------
// Small utility kernels
// ---------------------------------------------------------------------------
__global__ void cvt_f32_f16(const float* __restrict__ in, _Float16* __restrict__ out, int n) {
    int i = blockIdx.x * blockDim.x + threadIdx.x;
    if (i < n) out[i] = (_Float16)in[i];
}

// scale = gamma/sqrt(var+eps); bias = beta - mean*scale   (pv of bn_and_pad == bias)
__global__ void fold_bn1(const float* __restrict__ g, const float* __restrict__ be,
                         const float* __restrict__ me, const float* __restrict__ va,
                         float* __restrict__ s, float* __restrict__ b, int n) {
    int i = blockIdx.x * blockDim.x + threadIdx.x;
    if (i < n) {
        float sc = g[i] * rsqrtf(va[i] + EPSV);
        s[i] = sc;
        b[i] = be[i] - me[i] * sc;
    }
}

// composition of two BNs: y = bnB(bnA(v))
__global__ void fold_bn2(const float* gA, const float* bA, const float* mA, const float* vA,
                         const float* gB, const float* bB, const float* mB, const float* vB,
                         float* __restrict__ s, float* __restrict__ b, int n) {
    int i = blockIdx.x * blockDim.x + threadIdx.x;
    if (i < n) {
        float sA = gA[i] * rsqrtf(vA[i] + EPSV);
        float aA = bA[i] - mA[i] * sA;
        float sB = gB[i] * rsqrtf(vB[i] + EPSV);
        float aB = bB[i] - mB[i] * sB;
        s[i] = sA * sB;
        b[i] = aA * sB + aB;
    }
}

// ---------------------------------------------------------------------------
// LIF: mem = (mem+x)/2 ; spike = mem>=th ; mem *= (1-spike)    (scan over T=4)
// input layout (t,b,c,n) fp32 -> output spikes [C][M] fp16
// ---------------------------------------------------------------------------
__global__ void lif_from_tbcn(const float* __restrict__ in, _Float16* __restrict__ out,
                              int C, float thresh) {
    int t0 = blockIdx.x * blockDim.x + threadIdx.x;
    if (t0 >= C * 2048) return;
    int c = t0 >> 11, r = t0 & 2047;       // r = b*256 + n
    int b = r >> 8, n = r & 255;
    float mem = 0.f;
    for (int t = 0; t < 4; ++t) {
        float xv = in[(size_t)((t * 8 + b) * C + c) * 256 + n];
        mem = 0.5f * (mem + xv);
        float sp = (mem >= thresh) ? 1.f : 0.f;
        out[(size_t)c * MTOT + t * 2048 + r] = (_Float16)sp;
        mem *= (1.f - sp);
    }
}

// input [C][M] fp16 -> spikes [C][M] fp16
__global__ void lif_cm(const _Float16* __restrict__ in, _Float16* __restrict__ out,
                       int C, float thresh) {
    int t0 = blockIdx.x * blockDim.x + threadIdx.x;
    if (t0 >= C * 2048) return;
    int c = t0 >> 11, r = t0 & 2047;
    size_t base = (size_t)c * MTOT + r;
    float mem = 0.f;
    for (int t = 0; t < 4; ++t) {
        float xv = (float)in[base + t * 2048];
        mem = 0.5f * (mem + xv);
        float sp = (mem >= thresh) ? 1.f : 0.f;
        out[base + t * 2048] = (_Float16)sp;
        mem *= (1.f - sp);
    }
}

// ---------------------------------------------------------------------------
// Depthwise 3x3, border value pv[c] (= folded bnp bias).  [C][M] fp16 in/out.
// ---------------------------------------------------------------------------
__global__ void dwconv3x3(const _Float16* __restrict__ in, const float* __restrict__ wdw,
                          const float* __restrict__ pvb, _Float16* __restrict__ out) {
    int t = blockIdx.x * blockDim.x + threadIdx.x;   // 512*32*256 threads
    int n = t & 255, img = (t >> 8) & 31, c = t >> 13;
    int x = n & 15, y = n >> 4;
    const _Float16* base = in + (size_t)c * MTOT + img * 256;
    const float* wp = wdw + c * 9;
    float pv = pvb[c];
    float acc = 0.f;
    for (int ky = -1; ky <= 1; ++ky)
        for (int kx = -1; kx <= 1; ++kx) {
            int yy = y + ky, xx = x + kx;
            float v = (yy >= 0 && yy < 16 && xx >= 0 && xx < 16) ? (float)base[yy * 16 + xx] : pv;
            acc += v * wp[(ky + 1) * 3 + (kx + 1)];
        }
    out[(size_t)c * MTOT + img * 256 + n] = (_Float16)acc;
}

// ---------------------------------------------------------------------------
// Generic WMMA GEMM:  D[o][m] = affine_o( sum_k W[o][k] * X[k][m] )
// WG = 256 threads (8 waves), tile 128(O) x 64(M), K-step 32.
// A tile staged row-major [o][k] (A-frag = 2x ds_load_b128 per lane).
// B tile staged TRANSPOSED [m][k]  (B-frag = 2x ds_load_b128 per lane).
// Software-pipelined: next K-tile fetched to regs while WMMAs run.
// mode 0: out16[o*M+m] = val (fp16)
// mode 1: out32[((m>>8)*512+o)*256+(m&255)] = res32[idx] + val   (O==512)
// ---------------------------------------------------------------------------
__global__ void __launch_bounds__(256)
gemm_wmma(const _Float16* __restrict__ W, const _Float16* __restrict__ X,
          const float* __restrict__ scale, const float* __restrict__ bias,
          int O, int K, int mode,
          _Float16* __restrict__ out16,
          const float* __restrict__ res32, float* __restrict__ out32) {
    __shared__ __align__(16) _Float16 As[128 * 40];
    __shared__ __align__(16) _Float16 Bt[64 * 40];
    const int M = MTOT;
    int tid = threadIdx.x, lane = tid & 31, wave = tid >> 5;
    int wo = (wave & 3) * 32, wm = (wave >> 2) * 32;
    int otile0 = blockIdx.y * 128, mtile0 = blockIdx.x * 64;

    float8 acc[2][2];
    acc[0][0] = f8zero(); acc[0][1] = f8zero();
    acc[1][0] = f8zero(); acc[1][1] = f8zero();

    const int arow = tid >> 1, aseg = (tid & 1) * 16;   // A: 128 x 32
    const int brow = tid >> 3, bcol = (tid & 7) * 8;    // B: 32 x 64

    const int aoff = (lane & 16) ? 16 : 0;   // A-frag K chunk base (contiguous k = 16h + s)
    const int boff = (lane & 16) ? 16 : 0;   // B-frag K chunk base (contiguous k = 16h + s)

    // stage 0 global fetch
    const _Float16* wp = W + (size_t)(otile0 + arow) * K + aseg;
    const _Float16* xp = X + (size_t)brow * M + mtile0 + bcol;
    uint4 ra0 = *(const uint4*)wp;
    uint4 ra1 = *(const uint4*)(wp + 8);
    uint4 rb  = *(const uint4*)xp;

    for (int kb = 0; kb < K; kb += 32) {
        // commit staged regs to LDS
        *(uint4*)&As[arow * 40 + aseg]     = ra0;
        *(uint4*)&As[arow * 40 + aseg + 8] = ra1;
        {
            union { uint4 q; _Float16 h[8]; } u; u.q = rb;
            #pragma unroll
            for (int j = 0; j < 8; ++j) Bt[(bcol + j) * 40 + brow] = u.h[j];  // transpose
        }
        __syncthreads();

        // fetch next K-tile while this tile is consumed
        if (kb + 32 < K) {
            const _Float16* wpn = W + (size_t)(otile0 + arow) * K + (kb + 32) + aseg;
            const _Float16* xpn = X + (size_t)(kb + 32 + brow) * M + mtile0 + bcol;
            ra0 = *(const uint4*)wpn;
            ra1 = *(const uint4*)(wpn + 8);
            rb  = *(const uint4*)xpn;
        }
        if (kb + 64 < K) {
            __builtin_prefetch(W + (size_t)(otile0 + arow) * K + (kb + 64) + aseg, 0, 1);
            __builtin_prefetch(X + (size_t)(kb + 64 + brow) * M + mtile0 + bcol, 0, 1);
        }

        half16 af[2], bf[2];
        #pragma unroll
        for (int i = 0; i < 2; ++i) {
            int r = wo + i * 16 + (lane & 15);
            af[i] = ld_frag(&As[r * 40 + aoff], &As[r * 40 + aoff + 8]);
        }
        #pragma unroll
        for (int j = 0; j < 2; ++j) {
            int ccol = wm + j * 16 + (lane & 15);
            bf[j] = ld_frag(&Bt[ccol * 40 + boff], &Bt[ccol * 40 + boff + 8]);
        }
        #pragma unroll
        for (int i = 0; i < 2; ++i)
            #pragma unroll
            for (int j = 0; j < 2; ++j)
                acc[i][j] = __builtin_amdgcn_wmma_f32_16x16x32_f16(
                    false, af[i], false, bf[j], (short)0, acc[i][j], false, false);
        __syncthreads();
    }

    #pragma unroll
    for (int i = 0; i < 2; ++i)
        #pragma unroll
        for (int j = 0; j < 2; ++j) {
            int obase = otile0 + wo + i * 16 + ((lane & 16) ? 8 : 0);
            int m = mtile0 + wm + j * 16 + (lane & 15);
            #pragma unroll
            for (int v = 0; v < 8; ++v) {
                int o = obase + v;
                float val = acc[i][j][v] * scale[o] + bias[o];
                if (mode == 0) {
                    out16[(size_t)o * M + m] = (_Float16)val;
                } else {
                    size_t idx = ((size_t)(m >> 8) * 512 + o) * 256 + (m & 255);
                    out32[idx] = res32[idx] + val;
                }
            }
        }
}

// ---------------------------------------------------------------------------
// kv[t,b,h][d][e] = sum_n K[h*64+d][m0+n] * V[h*64+e][m0+n]    (64x64, K=256)
// 1 WG (128 thr / 4 waves) per (t,b,h); spike products are exact in fp16.
// Both tiles staged row-major [ch][n]: A-frag and B-frag chunks contiguous.
// ---------------------------------------------------------------------------
__global__ void __launch_bounds__(128)
kv_wmma(const _Float16* __restrict__ Kspk, const _Float16* __restrict__ Vspk,
        _Float16* __restrict__ KV) {
    __shared__ __align__(16) _Float16 Ka[64 * 40];
    __shared__ __align__(16) _Float16 Va[64 * 40];
    int g = blockIdx.x, tb = g >> 3, h = g & 7;
    size_t choff = (size_t)(h * 64) * MTOT + tb * 256;
    int tid = threadIdx.x, lane = tid & 31, wave = tid >> 5;
    float8 acc[4];
    #pragma unroll
    for (int j = 0; j < 4; ++j) acc[j] = f8zero();
    int row = tid >> 1, seg = (tid & 1) * 16;
    const int aoff = (lane & 16) ? 16 : 0;
    const int boff = (lane & 16) ? 16 : 0;

    for (int nb = 0; nb < 256; nb += 32) {
        const _Float16* kp = Kspk + choff + (size_t)row * MTOT + nb + seg;
        const _Float16* vp = Vspk + choff + (size_t)row * MTOT + nb + seg;
        *(uint4*)&Ka[row * 40 + seg]     = *(const uint4*)kp;
        *(uint4*)&Ka[row * 40 + seg + 8] = *(const uint4*)(kp + 8);
        *(uint4*)&Va[row * 40 + seg]     = *(const uint4*)vp;
        *(uint4*)&Va[row * 40 + seg + 8] = *(const uint4*)(vp + 8);
        __syncthreads();

        int r = wave * 16 + (lane & 15);
        half16 af = ld_frag(&Ka[r * 40 + aoff], &Ka[r * 40 + aoff + 8]);
        #pragma unroll
        for (int j = 0; j < 4; ++j) {
            int e = j * 16 + (lane & 15);
            half16 bf = ld_frag(&Va[e * 40 + boff], &Va[e * 40 + boff + 8]);
            acc[j] = __builtin_amdgcn_wmma_f32_16x16x32_f16(
                false, af, false, bf, (short)0, acc[j], false, false);
        }
        __syncthreads();
    }
    #pragma unroll
    for (int j = 0; j < 4; ++j)
        #pragma unroll
        for (int v = 0; v < 8; ++v) {
            int d = wave * 16 + v + ((lane & 16) ? 8 : 0);
            int e = j * 16 + (lane & 15);
            KV[(size_t)g * 4096 + d * 64 + e] = (_Float16)acc[j][v];
        }
}

// ---------------------------------------------------------------------------
// a^T[e][n] = 0.125 * sum_d kv[d][e] * q[n][d]   (64 x 256, K=64)
// A = kv^T staged [e][d]; B = Q staged TRANSPOSED [n][d] so fragments are
// contiguous ds_load_b128 pairs.
// ---------------------------------------------------------------------------
__global__ void __launch_bounds__(128)
attn_a_wmma(const _Float16* __restrict__ Qspk, const _Float16* __restrict__ KV,
            _Float16* __restrict__ Abuf) {
    __shared__ __align__(16) _Float16 Qt[256 * 72];   // [n][d]
    __shared__ __align__(16) _Float16 KT[64 * 72];    // [e][d]
    int g = blockIdx.x, tb = g >> 3, h = g & 7;
    size_t qoff = (size_t)(h * 64) * MTOT + tb * 256;
    int tid = threadIdx.x, lane = tid & 31, wave = tid >> 5;
    const int aoff = (lane & 16) ? 16 : 0;
    const int boff = (lane & 16) ? 16 : 0;

    // stage Q transposed: coalesced global b128 loads, scattered ds stores
    int d0 = tid >> 1, nbase = (tid & 1) * 128;
    for (int s8 = 0; s8 < 128; s8 += 8) {
        union { uint4 q; _Float16 h[8]; } u;
        u.q = *(const uint4*)(Qspk + qoff + (size_t)d0 * MTOT + nbase + s8);
        #pragma unroll
        for (int j = 0; j < 8; ++j) Qt[(nbase + s8 + j) * 72 + d0] = u.h[j];
    }
    // stage kv transposed
    int e0 = tid >> 1, dbase = (tid & 1) * 32;
    for (int dd = 0; dd < 32; ++dd)
        KT[e0 * 72 + dbase + dd] = KV[(size_t)g * 4096 + (dbase + dd) * 64 + e0];
    __syncthreads();

    for (int nt = 0; nt < 16; ++nt) {
        float8 acc = f8zero();
        #pragma unroll
        for (int kb = 0; kb < 64; kb += 32) {
            int r = wave * 16 + (lane & 15);
            half16 af = ld_frag(&KT[r * 72 + kb + aoff], &KT[r * 72 + kb + aoff + 8]);
            int ncol = nt * 16 + (lane & 15);
            half16 bf = ld_frag(&Qt[ncol * 72 + kb + boff], &Qt[ncol * 72 + kb + boff + 8]);
            acc = __builtin_amdgcn_wmma_f32_16x16x32_f16(
                false, af, false, bf, (short)0, acc, false, false);
        }
        #pragma unroll
        for (int v = 0; v < 8; ++v) {
            int ee = wave * 16 + v + ((lane & 16) ? 8 : 0);
            int n = nt * 16 + (lane & 15);
            Abuf[(size_t)(h * 64 + ee) * MTOT + tb * 256 + n] = (_Float16)(acc[v] * 0.125f);
        }
    }
}

// ---------------------------------------------------------------------------
// Host-side orchestration
// ---------------------------------------------------------------------------
extern "C" void kernel_launch(void* const* d_in, const int* in_sizes, int n_in,
                              void* d_out, int out_size, void* d_ws, size_t ws_size,
                              hipStream_t stream) {
    (void)in_sizes; (void)n_in; (void)out_size; (void)ws_size;
    const float* x = (const float*)d_in[0];
    char* ws = (char*)d_ws;
    const size_t MiB = 1u << 20;

    // fp16 weight pool
    _Float16* w16a[4]; _Float16* w16b[4];
    for (int i = 0; i < 4; ++i) {
        w16a[i] = (_Float16*)(ws + (size_t)i * MiB);
        w16b[i] = (_Float16*)(ws + (size_t)i * MiB + MiB / 2);
    }
    _Float16* mw1 = (_Float16*)(ws + 4 * MiB);     // 2048x512
    _Float16* mw2 = (_Float16*)(ws + 6 * MiB);     // 512x2048
    float* aff = (float*)(ws + 8 * MiB);
    _Float16* S0   = (_Float16*)(ws + 9 * MiB);    // spikes [512][8192]
    _Float16* Y1   = (_Float16*)(ws + 17 * MiB);
    _Float16* Zb   = (_Float16*)(ws + 25 * MiB);
    _Float16* Y2   = (_Float16*)(ws + 33 * MiB);
    _Float16* QKV[3] = { (_Float16*)(ws + 41 * MiB),
                         (_Float16*)(ws + 49 * MiB),
                         (_Float16*)(ws + 57 * MiB) };
    _Float16* KV   = (_Float16*)(ws + 65 * MiB);   // 256*64*64 fp16
    _Float16* Abuf = (_Float16*)(ws + 67 * MiB);
    _Float16* As   = (_Float16*)(ws + 75 * MiB);
    float*    X1   = (float*)   (ws + 83 * MiB);   // x + sdsa(x), fp32
    _Float16* Hpre = (_Float16*)(ws + 17 * MiB);   // overlays dead Y1..Qs (MLP phase)
    _Float16* Hspk = (_Float16*)(ws + 49 * MiB);   // overlays dead Ks..As (MLP phase)
    float* outp = (float*)d_out;

    // input leaf indices (insertion order): x, then per branch {w1, bnp(g,b,m,v),
    // wdw, wpw, bn(g,b,m,v)}, branch_bn(g,b,m,v); branches q,k,v,proj; then mlp.
    const int bases[4] = { 1, 16, 31, 46 };

    // weights -> fp16
    for (int i = 0; i < 4; ++i) {
        cvt_f32_f16<<<1024, 256, 0, stream>>>((const float*)d_in[bases[i] + 0], w16a[i], 512 * 512);
        cvt_f32_f16<<<1024, 256, 0, stream>>>((const float*)d_in[bases[i] + 6], w16b[i], 512 * 512);
    }
    cvt_f32_f16<<<4096, 256, 0, stream>>>((const float*)d_in[61], mw1, 2048 * 512);
    cvt_f32_f16<<<4096, 256, 0, stream>>>((const float*)d_in[66], mw2, 512 * 2048);

    // folded BN affines
    float* s1[4]; float* b1[4]; float* s2[4]; float* b2[4];
    for (int i = 0; i < 4; ++i) {
        int b = bases[i];
        s1[i] = aff + i * 2048;       b1[i] = s1[i] + 512;
        s2[i] = s1[i] + 1024;         b2[i] = s1[i] + 1536;
        fold_bn1<<<2, 256, 0, stream>>>((const float*)d_in[b + 1], (const float*)d_in[b + 2],
                                        (const float*)d_in[b + 3], (const float*)d_in[b + 4],
                                        s1[i], b1[i], 512);
        fold_bn2<<<2, 256, 0, stream>>>((const float*)d_in[b + 7], (const float*)d_in[b + 8],
                                        (const float*)d_in[b + 9], (const float*)d_in[b + 10],
                                        (const float*)d_in[b + 11], (const float*)d_in[b + 12],
                                        (const float*)d_in[b + 13], (const float*)d_in[b + 14],
                                        s2[i], b2[i], 512);
    }
    float* s3 = aff + 8192;  float* b3 = aff + 10240;
    float* s4 = aff + 12288; float* b4 = aff + 12800;
    fold_bn1<<<8, 256, 0, stream>>>((const float*)d_in[62], (const float*)d_in[63],
                                    (const float*)d_in[64], (const float*)d_in[65], s3, b3, 2048);
    fold_bn1<<<2, 256, 0, stream>>>((const float*)d_in[67], (const float*)d_in[68],
                                    (const float*)d_in[69], (const float*)d_in[70], s4, b4, 512);

    // spikes of x
    lif_from_tbcn<<<4096, 256, 0, stream>>>(x, S0, 512, 1.0f);

    dim3 g512(MTOT / 64, 4);    // O=512
    dim3 g2048(MTOT / 64, 16);  // O=2048

    // q, k, v branches: repconv + bn + lif
    for (int i = 0; i < 3; ++i) {
        gemm_wmma<<<g512, 256, 0, stream>>>(w16a[i], S0, s1[i], b1[i], 512, 512, 0,
                                            Y1, nullptr, nullptr);
        dwconv3x3<<<16384, 256, 0, stream>>>(Y1, (const float*)d_in[bases[i] + 5], b1[i], Zb);
        gemm_wmma<<<g512, 256, 0, stream>>>(w16b[i], Zb, s2[i], b2[i], 512, 512, 0,
                                            Y2, nullptr, nullptr);
        lif_cm<<<4096, 256, 0, stream>>>(Y2, QKV[i], 512, 1.0f);
    }

    // SDSA core
    kv_wmma<<<256, 128, 0, stream>>>(QKV[1], QKV[2], KV);
    attn_a_wmma<<<256, 128, 0, stream>>>(QKV[0], KV, Abuf);
    lif_cm<<<4096, 256, 0, stream>>>(Abuf, As, 512, 0.5f);

    // proj branch, residual fused: X1 = x + bn(repconv(As))
    gemm_wmma<<<g512, 256, 0, stream>>>(w16a[3], As, s1[3], b1[3], 512, 512, 0,
                                        Y1, nullptr, nullptr);
    dwconv3x3<<<16384, 256, 0, stream>>>(Y1, (const float*)d_in[51], b1[3], Zb);
    gemm_wmma<<<g512, 256, 0, stream>>>(w16b[3], Zb, s2[3], b2[3], 512, 512, 1,
                                        nullptr, x, X1);

    // MLP: out = X1 + bn2(W2 @ lif(bn1(W1 @ lif(X1))))
    lif_from_tbcn<<<4096, 256, 0, stream>>>(X1, S0, 512, 1.0f);
    gemm_wmma<<<g2048, 256, 0, stream>>>(mw1, S0, s3, b3, 2048, 512, 0,
                                         Hpre, nullptr, nullptr);
    lif_cm<<<16384, 256, 0, stream>>>(Hpre, Hspk, 2048, 1.0f);
    gemm_wmma<<<g512, 256, 0, stream>>>(mw2, Hspk, s4, b4, 512, 2048, 1,
                                        nullptr, X1, outp);
}
